// DETR3D_Core_53584011985464
// MI455X (gfx1250) — matrix operand order repaired and checked by reading, not compiled, not run on hardware
//
#include <hip/hip_runtime.h>
#include <math.h>

// ---------------------------------------------------------------------------
// Problem constants (from reference)
// ---------------------------------------------------------------------------
constexpr int Bk  = 2;      // batch
constexpr int NV  = 6;      // views
constexpr int Ck  = 256;    // channels == embed dim
constexpr int HH  = 116;
constexpr int WW  = 200;
constexpr int Qn  = 900;    // queries
constexpr int Dm  = 256;    // embed dim
constexpr int NHd = 8;      // heads
constexpr int HD  = 32;     // head dim (== WMMA K for f16!)
constexpr int TD  = 768;    // 3*D
constexpr int QP  = 928;    // Q padded to 32*29 = 16*58
constexpr int MP  = Bk * QP; // 1856 = 64*29 padded GEMM M

typedef _Float16 half_t;
typedef half_t v16h __attribute__((ext_vector_type(16)));
typedef half_t v8h  __attribute__((ext_vector_type(8)));
typedef float  v8f  __attribute__((ext_vector_type(8)));
typedef unsigned int u32x4 __attribute__((ext_vector_type(4)));
typedef int i32x4 __attribute__((ext_vector_type(4)));
typedef int i32x8 __attribute__((ext_vector_type(8)));

#if defined(__has_builtin)
#if __has_builtin(__builtin_amdgcn_tensor_load_to_lds)
#define HAVE_TDM 1
#endif
#endif

// ---------------------------------------------------------------------------
// Workspace layout (all 256B aligned)
// ---------------------------------------------------------------------------
constexpr size_t align256(size_t x) { return (x + 255) & ~size_t(255); }
constexpr size_t OFF_REF  = 0;                                                          // float[Qn*3]
constexpr size_t OFF_XH   = align256(OFF_REF  + sizeof(float)  * (size_t)Qn * 3);       // f16 [MP][Dm]
constexpr size_t OFF_WIN  = align256(OFF_XH   + sizeof(half_t) * (size_t)MP * Dm);      // f16 [TD][Dm]
constexpr size_t OFF_WOUT = align256(OFF_WIN  + sizeof(half_t) * (size_t)TD * Dm);      // f16 [Dm][Dm]
constexpr size_t OFF_QKV  = align256(OFF_WOUT + sizeof(half_t) * (size_t)Dm * Dm);      // f32 [MP][TD]
constexpr size_t OFF_QPK  = align256(OFF_QKV  + sizeof(float)  * (size_t)MP * TD);      // f16 [16][QP][HD]
constexpr size_t OFF_KPK  = align256(OFF_QPK  + sizeof(half_t) * (size_t)16 * QP * HD); // f16 [16][QP][HD]
constexpr size_t OFF_VT   = align256(OFF_KPK  + sizeof(half_t) * (size_t)16 * QP * HD); // f16 [16][HD][QP]
constexpr size_t OFF_OH   = align256(OFF_VT   + sizeof(half_t) * (size_t)16 * HD * QP); // f16 [MP][Dm]
constexpr size_t OFF_PROJ = align256(OFF_OH   + sizeof(half_t) * (size_t)MP * Dm);      // f32 [MP][Dm]

// ---------------------------------------------------------------------------
// WMMA helpers (CDNA5 16x16x32 f16 -> f32, layouts per ISA 7.12.2)
// ---------------------------------------------------------------------------
__device__ __forceinline__ v8f wmma32(v16h a, v16h b, v8f c) {
  // 8 args: (neg_a, A, neg_b, B, c_mod, C, reuse_a, reuse_b)
  return __builtin_amdgcn_wmma_f32_16x16x32_f16(false, a, false, b, (short)0, c,
                                                false, false);
}

// A-matrix fragment (16xK=32 f16): lane m = lane&15 holds row m; K-halves
// {8*kh + 0..7} and {16 + 8*kh + 0..7} with kh = lane>>4.  Two 16B loads.
__device__ __forceinline__ v16h load_a_frag(const half_t* base, int ld, int row0, int k0) {
  int lane = threadIdx.x & 31;
  int m = lane & 15, kh = lane >> 4;
  const half_t* p = base + (size_t)(row0 + m) * ld + k0 + 8 * kh;
  union { v16h v; v8h h[2]; } u;
  u.h[0] = *(const v8h*)(p);
  u.h[1] = *(const v8h*)(p + 16);
  return u.v;
}

// B-matrix fragment (K=32 x 16 f16) where the source is stored transposed,
// i.e. base is [N][K] row-major and B[k][n] = base[(n0+n)*ld + k0 + k].
// Lane n = lane&15 picks column n; K-halves {8*kh + 0..7} and
// {16 + 8*kh + 0..7} with kh = lane>>4 (matches A).  Two 16B loads.
__device__ __forceinline__ v16h load_b_frag(const half_t* base, int ld, int n0, int k0) {
  int lane = threadIdx.x & 31;
  int n = lane & 15, kh = lane >> 4;
  const half_t* p = base + (size_t)(n0 + n) * ld + k0 + 8 * kh;
  union { v16h v; v8h h[2]; } u;
  u.h[0] = *(const v8h*)(p);
  u.h[1] = *(const v8h*)(p + 16);
  return u.v;
}

// ---------------------------------------------------------------------------
// Tensor Data Mover: DMA a [rows=64][K] f16 tile (row-major, row stride = K
// elements) from global memory into LDS.  D# layout per ISA 08 sect. 8.
// ---------------------------------------------------------------------------
#ifdef HAVE_TDM
__device__ __forceinline__ void tdm_load_tile_64xK(unsigned int lds_off,
                                                   const half_t* gbase, int K) {
  unsigned long long ga = (unsigned long long)(uintptr_t)gbase;
  u32x4 g0;
  g0[0] = 1u;                                  // count=1, user mode, no gather
  g0[1] = lds_off;                             // lds_addr (bytes)
  g0[2] = (unsigned int)ga;                    // global_addr[31:0]
  g0[3] = (unsigned int)((ga >> 32) & 0x01FFFFFFu) | (2u << 30); // addr[56:32] | type=2

  i32x8 g1;
  g1[0] = (int)(1u << 16);                     // workgroup_mask=0, data_size=1 (2B)
  g1[1] = (int)(((unsigned)K & 0xFFFFu) << 16);             // tensor_dim0[15:0]
  g1[2] = (int)((((unsigned)K >> 16) & 0xFFFFu) | (64u << 16)); // dim0 hi | tensor_dim1 lo (=64)
  g1[3] = (int)(((unsigned)K & 0xFFFFu) << 16);             // dim1 hi(0) | tile_dim0 = K
  g1[4] = 64;                                  // tile_dim1 = 64 rows, tile_dim2 = 0
  g1[5] = K;                                   // tensor_dim0_stride[31:0] (elements)
  g1[6] = 0;                                   // stride hi | dim1_stride lo (unused, 2D)
  g1[7] = 0;

  i32x4 g2 = {0, 0, 0, 0};
  i32x4 g3 = {0, 0, 0, 0};
#if __clang_major__ >= 23
  i32x8 g4 = {0, 0, 0, 0, 0, 0, 0, 0};
  __builtin_amdgcn_tensor_load_to_lds(g0, g1, g2, g3, g4, 0);
#else
  __builtin_amdgcn_tensor_load_to_lds(g0, g1, g2, g3, 0);
#endif
}
#endif

// ---------------------------------------------------------------------------
// K1: ref = sigmoid(query_embedding @ ref_w.T + ref_b)   (Q only; batch-invariant)
// ---------------------------------------------------------------------------
__global__ void ref_kernel(const float* __restrict__ qe, const float* __restrict__ rw,
                           const float* __restrict__ rb, float* __restrict__ ref) {
  int q = blockIdx.x;
  int l = threadIdx.x;  // 32 threads
  float a0 = 0.f, a1 = 0.f, a2 = 0.f;
  for (int k = l; k < Dm; k += 32) {
    float x = qe[q * Dm + k];
    a0 += x * rw[0 * Dm + k];
    a1 += x * rw[1 * Dm + k];
    a2 += x * rw[2 * Dm + k];
  }
#pragma unroll
  for (int off = 16; off; off >>= 1) {
    a0 += __shfl_xor(a0, off, 32);
    a1 += __shfl_xor(a1, off, 32);
    a2 += __shfl_xor(a2, off, 32);
  }
  if (l == 0) {
    ref[q * 3 + 0] = 1.f / (1.f + __expf(-(a0 + rb[0])));
    ref[q * 3 + 1] = 1.f / (1.f + __expf(-(a1 + rb[1])));
    ref[q * 3 + 2] = 1.f / (1.f + __expf(-(a2 + rb[2])));
  }
}

// ---------------------------------------------------------------------------
// K2: project ref points, bilinear-sample all views, masked mean, add query,
//     write x as f16 (padded rows -> 0).   One block per (b, q_padded).
// ---------------------------------------------------------------------------
__global__ void sample_kernel(const float* __restrict__ feats,
                              const float* __restrict__ lidar,
                              const float* __restrict__ qe,
                              const float* __restrict__ ref,
                              half_t* __restrict__ xh) {
  int blk = blockIdx.x;
  int b = blk / QP, q = blk % QP;
  int c = threadIdx.x;  // 256 threads = channel
  size_t xrow = (size_t)(b * QP + q) * Dm;
  if (q >= Qn) { xh[xrow + c] = (half_t)0.f; return; }

  __shared__ float s_wx[NV], s_wy[NV], s_val[NV];
  __shared__ int   s_x0[NV], s_y0[NV];
  __shared__ float s_cnt;

  if (c < NV) {
    int n = c;
    float rx = ref[q * 3 + 0], ry = ref[q * 3 + 1], rz = ref[q * 3 + 2];
    const float* L = lidar + (size_t)(b * NV + n) * 16;
    float p0 = L[0] * rx + L[1] * ry + L[2]  * rz + L[3];
    float p1 = L[4] * rx + L[5] * ry + L[6]  * rz + L[7];
    float p2 = L[8] * rx + L[9] * ry + L[10] * rz + L[11];
    float denom = fabsf(p2) + 1e-5f;
    float pnx = (p0 / denom) / (float)(WW - 1) * 2.f - 1.f;
    float pny = (p1 / denom) / (float)(HH - 1) * 2.f - 1.f;
    bool valid = (p2 > 1e-5f) && (fmaxf(fabsf(pnx), fabsf(pny)) < 1.f);
    float ix = (pnx + 1.f) * (WW * 0.5f) - 0.5f;
    float iy = (pny + 1.f) * (HH * 0.5f) - 0.5f;
    float x0 = floorf(ix), y0 = floorf(iy);
    s_wx[n] = ix - x0; s_wy[n] = iy - y0;
    s_x0[n] = (int)x0; s_y0[n] = (int)y0;
    s_val[n] = valid ? 1.f : 0.f;
  }
  __syncthreads();
  if (c == 0) {
    float cnt = 0.f;
#pragma unroll
    for (int n = 0; n < NV; ++n) cnt += s_val[n];
    s_cnt = fmaxf(cnt, 1.f);
  }
  __syncthreads();

  float acc = 0.f;
  for (int n = 0; n < NV; ++n) {
    if (s_val[n] == 0.f) continue;  // uniform across block
    const float* fp = feats + ((size_t)(b * NV + n) * Ck + c) * (size_t)(HH * WW);
    int x0 = s_x0[n], y0 = s_y0[n];
    float wx1 = s_wx[n], wy1 = s_wy[n];
    // prefetch the cacheline we are about to gather (global_prefetch_b8)
    if (y0 >= 0 && y0 < HH && x0 >= 0 && x0 < WW)
      __builtin_prefetch(fp + (size_t)y0 * WW + x0, 0, 0);
    float v00 = 0.f, v01 = 0.f, v10 = 0.f, v11 = 0.f;
    if (y0 >= 0 && y0 < HH) {
      if (x0 >= 0     && x0 < WW)     v00 = fp[(size_t)y0 * WW + x0];
      if (x0 + 1 >= 0 && x0 + 1 < WW) v01 = fp[(size_t)y0 * WW + x0 + 1];
    }
    if (y0 + 1 >= 0 && y0 + 1 < HH) {
      if (x0 >= 0     && x0 < WW)     v10 = fp[(size_t)(y0 + 1) * WW + x0];
      if (x0 + 1 >= 0 && x0 + 1 < WW) v11 = fp[(size_t)(y0 + 1) * WW + x0 + 1];
    }
    acc += v00 * (1.f - wx1) * (1.f - wy1) + v01 * wx1 * (1.f - wy1) +
           v10 * (1.f - wx1) * wy1         + v11 * wx1 * wy1;
  }
  float xv = qe[q * Dm + c] + acc / s_cnt;
  xh[xrow + c] = (half_t)xv;
}

// ---------------------------------------------------------------------------
// K3: pack in_proj_w and out_proj_w to f16
// ---------------------------------------------------------------------------
__global__ void pack_weights_kernel(const float* __restrict__ win,
                                    const float* __restrict__ wout,
                                    half_t* __restrict__ winh,
                                    half_t* __restrict__ wouth) {
  int i = blockIdx.x * blockDim.x + threadIdx.x;
  constexpr int N1 = TD * Dm;      // 196608
  constexpr int N2 = Dm * Dm;      // 65536
  if (i < N1) winh[i] = (half_t)win[i];
  else if (i < N1 + N2) { int j = i - N1; wouth[j] = (half_t)wout[j]; }
}

// ---------------------------------------------------------------------------
// K4/K7: WMMA GEMM  C[M][N] = A[M][K](f16) * Wt[N][K](f16)^T + bias
//  block = 128 threads (4 waves), tile 64x64, each wave does a 16x64 strip.
//  The 64xK weight tile is DMA'd into LDS once per block by the Tensor Data
//  Mover (wave 0 issues tensor_load_to_lds, waits TENSORcnt, then barrier);
//  all B-fragments then come from LDS.  K must be 256.
// ---------------------------------------------------------------------------
__global__ void gemm_f16_kernel(const half_t* __restrict__ A,
                                const half_t* __restrict__ Wt,
                                const float* __restrict__ bias,
                                float* __restrict__ C, int N, int K) {
  __shared__ half_t sB[64 * 256];  // 32 KB: 64 weight rows x K halves
  int wave = threadIdx.x >> 5;
  int lane = threadIdx.x & 31;
  int row0 = blockIdx.y * 64 + wave * 16;
  int col0 = blockIdx.x * 64;

#ifdef HAVE_TDM
  if (threadIdx.x < 32) {   // wave 0 drives the TDM; EXEC is ignored by TDM
    tdm_load_tile_64xK((unsigned int)(uintptr_t)&sB[0],
                       Wt + (size_t)col0 * K, K);
    __builtin_amdgcn_s_wait_tensorcnt(0);
  }
  __syncthreads();
#else
  for (int i = threadIdx.x; i < 64 * (K / 8); i += blockDim.x) {
    int r = i / (K / 8), cc = i % (K / 8);
    *(v8h*)(sB + (size_t)r * K + cc * 8) =
        *(const v8h*)(Wt + (size_t)(col0 + r) * K + cc * 8);
  }
  __syncthreads();
#endif

  v8f acc[4] = {v8f{}, v8f{}, v8f{}, v8f{}};
  for (int k0 = 0; k0 < K; k0 += 32) {
    v16h a = load_a_frag(A, K, row0, k0);
#pragma unroll
    for (int t = 0; t < 4; ++t) {
      v16h bfrag = load_b_frag(sB, K, 16 * t, k0);  // LDS ds_loads
      acc[t] = wmma32(a, bfrag, acc[t]);
    }
  }
  int kh = lane >> 4, nn = lane & 15;
#pragma unroll
  for (int t = 0; t < 4; ++t) {
    int col = col0 + 16 * t + nn;
    float bval = bias[col];
#pragma unroll
    for (int r = 0; r < 8; ++r) {
      int row = row0 + r + 8 * kh;
      C[(size_t)row * N + col] = acc[t][r] + bval;
    }
  }
}

// ---------------------------------------------------------------------------
// K5: split qkv (f32) into f16 head-layout buffers, fold 1/sqrt(hd) into q,
//     write V transposed [bh][d][key] so attention B-frags are contiguous.
// ---------------------------------------------------------------------------
__global__ void qkv_pack_kernel(const float* __restrict__ qkv,
                                half_t* __restrict__ qp,
                                half_t* __restrict__ kp,
                                half_t* __restrict__ vT) {
  int row = blockIdx.x;  // 0..MP-1
  int b = row / QP, q = row % QP;
  for (int n = threadIdx.x; n < TD; n += blockDim.x) {
    float v = (q < Qn) ? qkv[(size_t)row * TD + n] : 0.f;
    int sec = n >> 8;         // 0=q 1=k 2=v
    int d256 = n & 255;
    int head = d256 >> 5, d = d256 & 31;
    int bh = b * NHd + head;
    if (sec == 0)
      qp[((size_t)bh * QP + q) * HD + d] = (half_t)(v * 0.1767766953f); // 1/sqrt(32)
    else if (sec == 1)
      kp[((size_t)bh * QP + q) * HD + d] = (half_t)v;
    else
      vT[((size_t)bh * HD + d) * QP + q] = (half_t)v;
  }
}

// ---------------------------------------------------------------------------
// K6: flash attention.  grid (QP/16, B*NH), block = 1 wave (32 threads).
//     hd = 32 == WMMA K; scores tile 16x32 keys per iteration (2 WMMAs),
//     online softmax, P through LDS, O += P*V (2 WMMAs).
// ---------------------------------------------------------------------------
__global__ void attn_kernel(const half_t* __restrict__ qp,
                            const half_t* __restrict__ kp,
                            const half_t* __restrict__ vT,
                            half_t* __restrict__ oh) {
  int bh = blockIdx.y;
  int b = bh >> 3, h = bh & 7;
  int row0 = blockIdx.x * 16;
  int lane = threadIdx.x;
  int nn = lane & 15, kh = lane >> 4;

  const half_t* qb = qp + (size_t)bh * QP * HD;
  const half_t* kb = kp + (size_t)bh * QP * HD;
  const half_t* vb = vT + (size_t)bh * HD * QP;

  __shared__ half_t sP[16 * 32];

  v16h qa = load_a_frag(qb, HD, row0, 0);
  v8f O0 = {}, O1 = {};
  float mrow[8], lrow[8];
#pragma unroll
  for (int r = 0; r < 8; ++r) { mrow[r] = -3.0e38f; lrow[r] = 0.f; }

  for (int k0 = 0; k0 < QP; k0 += 32) {
    v16h kb0 = load_b_frag(kb, HD, k0, 0);       // keys k0..k0+15
    v16h kb1 = load_b_frag(kb, HD, k0 + 16, 0);  // keys k0+16..k0+31
    v8f z = {};
    v8f S0 = wmma32(qa, kb0, z);
    v8f S1 = wmma32(qa, kb1, z);
    bool msk0 = (k0 + nn) >= Qn;
    bool msk1 = (k0 + 16 + nn) >= Qn;
#pragma unroll
    for (int r = 0; r < 8; ++r) {
      float s0 = msk0 ? -3.0e38f : S0[r];
      float s1 = msk1 ? -3.0e38f : S1[r];
      float tmax = fmaxf(s0, s1);
#pragma unroll
      for (int off = 8; off; off >>= 1) tmax = fmaxf(tmax, __shfl_xor(tmax, off, 16));
      float mn = fmaxf(mrow[r], tmax);
      float corr = __expf(mrow[r] - mn);
      float p0 = __expf(s0 - mn);
      float p1 = __expf(s1 - mn);
      float rs = p0 + p1;
#pragma unroll
      for (int off = 8; off; off >>= 1) rs += __shfl_xor(rs, off, 16);
      lrow[r] = lrow[r] * corr + rs;
      mrow[r] = mn;
      O0[r] *= corr;
      O1[r] *= corr;
      int prow = r + 8 * kh;
      sP[prow * 32 + nn]      = (half_t)p0;
      sP[prow * 32 + 16 + nn] = (half_t)p1;
    }
    __syncthreads();  // single-wave block: cheap, orders LDS P-tile
    v16h pa  = load_a_frag(sP, 32, 0, 0);          // P: 16 rows x 32 keys
    v16h vb0 = load_b_frag(vb, QP, 0,  k0);        // V: 32 keys x d 0..15
    v16h vb1 = load_b_frag(vb, QP, 16, k0);        // V: 32 keys x d 16..31
    O0 = wmma32(pa, vb0, O0);
    O1 = wmma32(pa, vb1, O1);
    __syncthreads();
  }

#pragma unroll
  for (int r = 0; r < 8; ++r) {
    int qrow = row0 + r + 8 * kh;
    if (qrow < Qn) {
      float inv = 1.f / lrow[r];
      size_t base = (size_t)(b * QP + qrow) * Dm + h * HD;
      oh[base + nn]      = (half_t)(O0[r] * inv);
      oh[base + 16 + nn] = (half_t)(O1[r] * inv);
    }
  }
}

// ---------------------------------------------------------------------------
// K8: LayerNorm + cls/box heads.  One block per (b,q), 256 threads (8 waves).
// ---------------------------------------------------------------------------
__global__ void ln_heads_kernel(const float* __restrict__ proj,
                                const float* __restrict__ nw, const float* __restrict__ nb,
                                const float* __restrict__ cls_w, const float* __restrict__ cls_b,
                                const float* __restrict__ box_w, const float* __restrict__ box_b,
                                float* __restrict__ out) {
  int blk = blockIdx.x;
  int b = blk / Qn, q = blk % Qn;
  int t = threadIdx.x;
  const float* row = proj + (size_t)(b * QP + q) * Dm;

  __shared__ float warp_red[8];
  __shared__ float s_mu, s_rstd;
  __shared__ float sx[Dm];

  float v = row[t];
  float s = v;
#pragma unroll
  for (int off = 16; off; off >>= 1) s += __shfl_xor(s, off, 32);
  if ((t & 31) == 0) warp_red[t >> 5] = s;
  __syncthreads();
  if (t == 0) {
    float tot = 0.f;
#pragma unroll
    for (int i = 0; i < 8; ++i) tot += warp_red[i];
    s_mu = tot / (float)Dm;
  }
  __syncthreads();
  float d = v - s_mu;
  float s2 = d * d;
#pragma unroll
  for (int off = 16; off; off >>= 1) s2 += __shfl_xor(s2, off, 32);
  if ((t & 31) == 0) warp_red[t >> 5] = s2;
  __syncthreads();
  if (t == 0) {
    float tot = 0.f;
#pragma unroll
    for (int i = 0; i < 8; ++i) tot += warp_red[i];
    s_rstd = rsqrtf(tot / (float)Dm + 1e-5f);
  }
  __syncthreads();
  sx[t] = d * s_rstd * nw[t] + nb[t];
  __syncthreads();

  int w = t >> 5, l = t & 31;  // 8 waves -> 8 outputs (2 cls + 6 box)
  const float* Wrow;
  float bval;
  bool sig;
  if (w < 2) { Wrow = cls_w + w * Dm; bval = cls_b[w]; sig = false; }
  else       { int j = w - 2; Wrow = box_w + j * Dm; bval = box_b[j]; sig = (j < 3); }
  float acc = 0.f;
  for (int kk = l; kk < Dm; kk += 32) acc += sx[kk] * Wrow[kk];
#pragma unroll
  for (int off = 16; off; off >>= 1) acc += __shfl_xor(acc, off, 32);
  if (l == 0) {
    float o = acc + bval;
    if (sig) o = 1.f / (1.f + __expf(-o));
    if (w < 2) out[(size_t)(b * Qn + q) * 2 + w] = o;                        // cls
    else       out[(size_t)(Bk * Qn * 2) + (size_t)(b * Qn + q) * 6 + (w - 2)] = o; // box
  }
}

// ---------------------------------------------------------------------------
// Launcher
// ---------------------------------------------------------------------------
extern "C" void kernel_launch(void* const* d_in, const int* in_sizes, int n_in,
                              void* d_out, int out_size, void* d_ws, size_t ws_size,
                              hipStream_t stream) {
  const float* feats  = (const float*)d_in[0];
  const float* lidar  = (const float*)d_in[1];
  const float* qe     = (const float*)d_in[2];
  const float* ref_w  = (const float*)d_in[3];
  const float* ref_b  = (const float*)d_in[4];
  const float* in_w   = (const float*)d_in[5];
  const float* in_b   = (const float*)d_in[6];
  const float* out_w  = (const float*)d_in[7];
  const float* out_b  = (const float*)d_in[8];
  const float* norm_w = (const float*)d_in[9];
  const float* norm_b = (const float*)d_in[10];
  const float* cls_w  = (const float*)d_in[11];
  const float* cls_b  = (const float*)d_in[12];
  const float* box_w  = (const float*)d_in[13];
  const float* box_b  = (const float*)d_in[14];
  float* outp = (float*)d_out;

  char* ws = (char*)d_ws;
  float*  refp  = (float*)(ws + OFF_REF);
  half_t* xh    = (half_t*)(ws + OFF_XH);
  half_t* winh  = (half_t*)(ws + OFF_WIN);
  half_t* wouth = (half_t*)(ws + OFF_WOUT);
  float*  qkv   = (float*)(ws + OFF_QKV);
  half_t* qpk   = (half_t*)(ws + OFF_QPK);
  half_t* kpk   = (half_t*)(ws + OFF_KPK);
  half_t* vT    = (half_t*)(ws + OFF_VT);
  half_t* oh    = (half_t*)(ws + OFF_OH);
  float*  projp = (float*)(ws + OFF_PROJ);

  // K1: reference points (sigmoid projection of query embedding)
  ref_kernel<<<Qn, 32, 0, stream>>>(qe, ref_w, ref_b, refp);

  // K3: weight f32->f16 packs (independent of K1/K2)
  pack_weights_kernel<<<(TD * Dm + Dm * Dm + 255) / 256, 256, 0, stream>>>(
      in_w, out_w, winh, wouth);

  // K2: projection + bilinear sampling + masked mean + residual -> x (f16)
  sample_kernel<<<Bk * QP, 256, 0, stream>>>(feats, lidar, qe, refp, xh);

  // K4: qkv = x @ in_proj_w^T + b   (WMMA GEMM + TDM weight staging)
  gemm_f16_kernel<<<dim3(TD / 64, MP / 64), 128, 0, stream>>>(xh, winh, in_b, qkv, TD, Dm);

  // K5: split into per-head f16 q/k/vT (scale folded into q)
  qkv_pack_kernel<<<MP, 256, 0, stream>>>(qkv, qpk, kpk, vT);

  // K6: flash attention (WMMA), 16 (b,h) x 58 q-tiles
  attn_kernel<<<dim3(QP / 16, Bk * NHd), 32, 0, stream>>>(qpk, kpk, vT, oh);

  // K7: out_proj GEMM (WMMA + TDM weight staging, M=1856 N=256 K=256)
  gemm_f16_kernel<<<dim3(Dm / 64, MP / 64), 128, 0, stream>>>(oh, wouth, out_b, projp, Dm, Dm);

  // K8: LayerNorm + cls/box heads -> d_out (cls flat then box flat)
  ln_heads_kernel<<<Bk * Qn, 256, 0, stream>>>(projp, norm_w, norm_b,
                                               cls_w, cls_b, box_w, box_b, outp);
}
